// SelfAttention_66649302499798
// MI455X (gfx1250) — matrix operand order repaired and checked by reading, not compiled, not run on hardware
//
#include <hip/hip_runtime.h>
#include <hip/hip_bf16.h>

// ---------------------------------------------------------------------------
// Self-attention (B=4, S=2048, D=1024), fp32 in/out, f16 WMMA internally.
// Stages: (1) convert X->f16, W->f16 transposed; (2) QKV projection GEMMs via
// v_wmma_f32_16x16x32_f16 (V stored transposed); (3) flash-attention with
// online softmax, 8 waves/workgroup, scores + P exchanged through LDS.
// Round 4: proj widened to 32x64 tiles per wave (2 A row-sets share B frags:
// 1.5 loads/WMMA instead of 2.5); single base pointer per stream with all
// tile displacements folded into 24-bit instruction offsets.
// ---------------------------------------------------------------------------

constexpr int D_IN  = 1024;
constexpr int D_OUT = 1024;
constexpr int SEQ   = 2048;
constexpr int BATCH = 4;

typedef __attribute__((ext_vector_type(16))) _Float16 v16h;
typedef __attribute__((ext_vector_type(8)))  _Float16 v8h;
typedef __attribute__((ext_vector_type(8)))  float    v8f;
typedef __attribute__((ext_vector_type(4)))  int      v4i;

#define AS1 __attribute__((address_space(1)))
#define AS3 __attribute__((address_space(3)))

union FragU { v16h v; v8h h[2]; };

// A fragment (16x32 f16, MxK): lane row = lane%16; halfs 0..7 = K hi*8+0..7,
// halfs 8..15 = K 16+hi*8+0..7.  `rowptr` points at A[row][kbase].
__device__ __forceinline__ v16h load_a_frag(const _Float16* rowptr, int hi) {
    FragU f;
    f.h[0] = *(const v8h*)(rowptr + hi * 8);
    f.h[1] = *(const v8h*)(rowptr + 16 + hi * 8);
    return f.v;
}

// B fragment (32x16 f16, KxN): lane col = lane%16; halfs 0..7 = K hi*8+0..7,
// halfs 8..15 = K 16+hi*8+0..7.
// `colptr` points at B[kbase][col] laid out contiguously over K (i.e. the
// K-major / transposed storage of the logical B matrix).
__device__ __forceinline__ v16h load_b_frag(const _Float16* colptr, int hi) {
    FragU f;
    f.h[0] = *(const v8h*)(colptr + hi * 8);
    f.h[1] = *(const v8h*)(colptr + 16 + hi * 8);
    return f.v;
}

__device__ __forceinline__ v8f wmma_f16(v16h a, v16h b, v8f c) {
    return __builtin_amdgcn_wmma_f32_16x16x32_f16(false, a, false, b,
                                                  (short)0, c, false, false);
}

// ---------------------------------------------------------------------------
// Stage 1a: fp32 -> f16 for X
__global__ void cvt_x_kernel(const float* __restrict__ X,
                             _Float16* __restrict__ Xh, int n) {
    int i = blockIdx.x * blockDim.x + threadIdx.x;
    if (i < n) Xh[i] = (_Float16)X[i];
}

// Stage 1b: fp32 [D_IN][D_OUT] -> f16 transposed [D_OUT][D_IN], 3 weights.
__global__ void cvt_w_kernel(const float* __restrict__ Wq,
                             const float* __restrict__ Wk,
                             const float* __restrict__ Wv,
                             _Float16* __restrict__ Wt) {
    int z = blockIdx.z;
    const float* W = (z == 0) ? Wq : (z == 1) ? Wk : Wv;
    int idx = blockIdx.x * blockDim.x + threadIdx.x;   // over D_OUT*D_IN
    int e = idx / D_IN;
    int d = idx % D_IN;
    Wt[(size_t)z * D_OUT * D_IN + idx] = (_Float16)W[(size_t)d * D_OUT + e];
}

// ---------------------------------------------------------------------------
// Stage 2: projections.  One wave per 32x64 output tile: two 16-row A sets
// share the four B (weight) fragments -> 16 WMMAs per 64-k step at 1.5
// b128-loads/WMMA.  Ping-pong buffers; all intra-tile displacements (row-set
// +16 rows = 32768 B, weight column-tile t*16 rows = t*32768 B) are folded
// into instruction offsets off a single base pointer per stream.  The final
// refill over-reads one k-step into adjacent workspace (valid, unused).
// z=0 -> Q row-major, z=1 -> K row-major, z=2 -> V transposed [b][e][s].
__global__ __launch_bounds__(32)
void proj_kernel(const _Float16* __restrict__ Xh,   // [B*S][D_IN]
                 const _Float16* __restrict__ Wt,   // [3][D_OUT][D_IN]
                 _Float16* __restrict__ Qh,
                 _Float16* __restrict__ Kh,
                 _Float16* __restrict__ Vt) {
    const int lane = threadIdx.x & 31;
    const int l15  = lane & 15;
    const int hi   = lane >> 4;
    const int r0   = blockIdx.x * 32;   // rows in [0, B*S)
    const int c0   = blockIdx.y * 64;   // cols in [0, D_OUT)
    const int z    = blockIdx.z;

    constexpr int R16 = 16 * D_IN;      // +16 rows, in halfs (32768 bytes)

    const _Float16* ap = Xh + (size_t)(r0 + l15) * D_IN;
    const _Float16* bp = Wt + (size_t)z * D_OUT * D_IN + (size_t)(c0 + l15) * D_IN;

    v8f acc[2][4] = {};

    // Prologue: buffer 0 <- k = 0.
    v16h a0[2], b0[4], a1[2], b1[4];
    a0[0] = load_a_frag(ap, hi);
    a0[1] = load_a_frag(ap + R16, hi);
#pragma unroll
    for (int t = 0; t < 4; ++t) b0[t] = load_b_frag(bp + t * R16, hi);

    for (int k = 0; k < D_IN; k += 64) {
        // buffer 1 <- k+32 (in flight while buffer 0 is consumed)
        a1[0] = load_a_frag(ap + k + 32, hi);
        a1[1] = load_a_frag(ap + R16 + k + 32, hi);
#pragma unroll
        for (int t = 0; t < 4; ++t) b1[t] = load_b_frag(bp + t * R16 + k + 32, hi);
#pragma unroll
        for (int t = 0; t < 4; ++t) acc[0][t] = wmma_f16(a0[0], b0[t], acc[0][t]);
#pragma unroll
        for (int t = 0; t < 4; ++t) acc[1][t] = wmma_f16(a0[1], b0[t], acc[1][t]);
        // buffer 0 <- k+64 (in flight while buffer 1 is consumed)
        a0[0] = load_a_frag(ap + k + 64, hi);
        a0[1] = load_a_frag(ap + R16 + k + 64, hi);
#pragma unroll
        for (int t = 0; t < 4; ++t) b0[t] = load_b_frag(bp + t * R16 + k + 64, hi);
#pragma unroll
        for (int t = 0; t < 4; ++t) acc[0][t] = wmma_f16(a1[0], b1[t], acc[0][t]);
#pragma unroll
        for (int t = 0; t < 4; ++t) acc[1][t] = wmma_f16(a1[1], b1[t], acc[1][t]);
    }

#pragma unroll
    for (int rs = 0; rs < 2; ++rs) {
#pragma unroll
        for (int t = 0; t < 4; ++t) {
#pragma unroll
            for (int r = 0; r < 8; ++r) {
                int row = r0 + rs * 16 + r + 8 * hi;
                int col = c0 + t * 16 + l15;
                _Float16 hv = (_Float16)acc[rs][t][r];
                if (z == 0) {
                    Qh[(size_t)row * D_OUT + col] = hv;
                } else if (z == 1) {
                    Kh[(size_t)row * D_OUT + col] = hv;
                } else {
                    int b = row >> 11;          // row / SEQ
                    int s = row & (SEQ - 1);
                    Vt[((size_t)b * D_OUT + col) * SEQ + s] = hv;
                }
            }
        }
    }
}

// ---------------------------------------------------------------------------
// Stage 3: flash attention.  Workgroup = (q-tile of 16, batch), 8 waves.
// Per 64-key block: waves 0..7 = (ntile = w&3, khalf = w>>2) compute the
// 16x64 score tile cooperatively; lanes 0..15 of wave 0 run the online
// softmax; every wave owns a 16x128 e-slice of the output accumulator.
__global__ __launch_bounds__(256)
void attn_kernel(const _Float16* __restrict__ Qh,   // [B*S][D]
                 const _Float16* __restrict__ Kh,   // [B*S][D]
                 const _Float16* __restrict__ Vt,   // [B][D][S]
                 float* __restrict__ out) {         // [B*S][D] fp32
    __shared__ __align__(32) _Float16 sQ[16 * D_IN];   // 32 KB
    __shared__ __align__(32) float    sS[16 * 64];     // scores, 4 KB
    __shared__ __align__(32) float    sPart[4][256];   // k-half partials, 4 KB
    __shared__ __align__(32) _Float16 sP[16 * 64];     // probs (f16), 2 KB
    __shared__ float sM[16], sL[16], sAlpha[16];

    const int tid   = threadIdx.x;
    const int lane  = tid & 31;
    const int w     = tid >> 5;
    const int l15   = lane & 15;
    const int hi    = lane >> 4;
    const int q0    = blockIdx.x * 16;
    const int b     = blockIdx.y;
    const int ntile = w & 3;
    const int khalf = w >> 2;
    const int ebase = w * 128;
    const float scale = 0.03125f;   // 1/sqrt(1024)

    // Q tile (rows q0..q0+15 of batch b) is contiguous in Qh.  Prefer the
    // CDNA5 async global->LDS copy (ASYNCcnt) when the builtin exists.
    {
        const _Float16* qsrc = Qh + ((size_t)b * SEQ + q0) * D_IN;
#if __has_builtin(__builtin_amdgcn_global_load_async_to_lds_b128)
        for (int i = tid * 8; i < 16 * D_IN; i += 256 * 8) {
            __builtin_amdgcn_global_load_async_to_lds_b128(
                (AS1 v4i*)(v4i*)(qsrc + i), (AS3 v4i*)(v4i*)(sQ + i), 0, 0);
        }
#if __has_builtin(__builtin_amdgcn_s_wait_asynccnt)
        __builtin_amdgcn_s_wait_asynccnt(0);
#else
        asm volatile("s_wait_asynccnt 0" ::: "memory");
#endif
#else
        for (int i = tid * 8; i < 16 * D_IN; i += 256 * 8)
            *(v8h*)(sQ + i) = *(const v8h*)(qsrc + i);
#endif
    }
    if (tid < 16) { sM[tid] = -1e30f; sL[tid] = 0.0f; }
    __syncthreads();

    v8f acc[8] = {};

    // Per-wave V base: column e = ebase + l15, key offset advances with kb;
    // the t*16-column displacement (t*65536 B) folds into the instr offset.
    const _Float16* vbase = Vt + ((size_t)b * D_OUT + ebase + l15) * SEQ;

    for (int kb = 0; kb < SEQ; kb += 64) {
        // (a) partial scores: this wave's 16x16 tile, half of the D
        // reduction.  Ping-pong buffers + two independent accumulator
        // chains so WMMAs never serialize on C or wait on register copies.
        v8f c;
        {
            const _Float16* qrow = sQ + (size_t)l15 * D_IN + khalf * 512;
            const _Float16* kcol =
                Kh + ((size_t)b * SEQ + kb + ntile * 16 + l15) * D_IN + khalf * 512;
            v8f c0 = {}, c1 = {};
            v16h a0 = load_a_frag(qrow, hi);
            v16h f0 = load_b_frag(kcol, hi);
            for (int kk = 0; kk < 512; kk += 64) {
                v16h a1 = load_a_frag(qrow + kk + 32, hi);
                v16h f1 = load_b_frag(kcol + kk + 32, hi);
                c0 = wmma_f16(a0, f0, c0);
                a0 = load_a_frag(qrow + kk + 64, hi);
                f0 = load_b_frag(kcol + kk + 64, hi);
                c1 = wmma_f16(a1, f1, c1);
            }
            c = c0 + c1;
        }
        if (khalf == 1) {
#pragma unroll
            for (int r = 0; r < 8; ++r) sPart[ntile][r * 32 + lane] = c[r];
        }
        __syncthreads();
        if (khalf == 0) {
#pragma unroll
            for (int r = 0; r < 8; ++r) {
                float v = (c[r] + sPart[ntile][r * 32 + lane]) * scale;
                sS[(r + 8 * hi) * 64 + ntile * 16 + l15] = v;
            }
        }
        __syncthreads();

        // (b) online softmax: one lane per query row.
        if (tid < 16) {
            float mold = sM[tid];
            float mx   = mold;
            for (int j = 0; j < 64; ++j) mx = fmaxf(mx, sS[tid * 64 + j]);
            float sum = 0.0f;
            for (int j = 0; j < 64; ++j) {
                float p = __expf(sS[tid * 64 + j] - mx);
                sum += p;
                sP[tid * 64 + j] = (_Float16)p;
            }
            float alpha = __expf(mold - mx);
            sAlpha[tid] = alpha;
            sM[tid]     = mx;
            sL[tid]     = sL[tid] * alpha + sum;
        }
        __syncthreads();

        // (c) rescale accumulators, then acc += P(16x64) @ V(64 x 128-slice).
        {
            float al[8];
#pragma unroll
            for (int r = 0; r < 8; ++r) al[r] = sAlpha[r + 8 * hi];
#pragma unroll
            for (int t = 0; t < 8; ++t)
#pragma unroll
                for (int r = 0; r < 8; ++r) acc[t][r] *= al[r];
        }
#pragma unroll
        for (int kk = 0; kk < 64; kk += 32) {
            // Batch all independent loads before the WMMA chain so the wave
            // waits once with many requests in flight, not per-WMMA.
            v16h a = load_a_frag(sP + l15 * 64 + kk, hi);
            v16h bf[8];
#pragma unroll
            for (int t = 0; t < 8; ++t)
                bf[t] = load_b_frag(vbase + (size_t)t * 16 * SEQ + kb + kk, hi);
#pragma unroll
            for (int t = 0; t < 8; ++t) acc[t] = wmma_f16(a, bf[t], acc[t]);
        }
        __syncthreads();
    }

    // Epilogue: normalize by running denominator, write fp32 output.
    float li[8];
#pragma unroll
    for (int r = 0; r < 8; ++r) li[r] = 1.0f / sL[r + 8 * hi];
#pragma unroll
    for (int t = 0; t < 8; ++t) {
#pragma unroll
        for (int r = 0; r < 8; ++r) {
            size_t row = (size_t)b * SEQ + q0 + r + 8 * hi;
            int    col = ebase + t * 16 + l15;
            out[row * D_OUT + col] = acc[t][r] * li[r];
        }
    }
}

// ---------------------------------------------------------------------------
extern "C" void kernel_launch(void* const* d_in, const int* in_sizes, int n_in,
                              void* d_out, int out_size, void* d_ws, size_t ws_size,
                              hipStream_t stream) {
    const float* X  = (const float*)d_in[0];
    const float* Wq = (const float*)d_in[1];
    const float* Wk = (const float*)d_in[2];
    const float* Wv = (const float*)d_in[3];
    float* out = (float*)d_out;

    const size_t nX = (size_t)BATCH * SEQ * D_IN;      // 8,388,608
    const size_t nW = (size_t)D_IN * D_OUT;            // 1,048,576

    // Workspace layout (f16): Xh | Wt[3] | Qh | Kh | Vt  -> ~70 MB total.
    _Float16* Xh = (_Float16*)d_ws;
    _Float16* Wt = Xh + nX;
    _Float16* Qh = Wt + 3 * nW;
    _Float16* Kh = Qh + nX;
    _Float16* Vt = Kh + nX;

    // Stage 1: conversions.
    cvt_x_kernel<<<dim3((unsigned)(nX / 256)), dim3(256), 0, stream>>>(X, Xh, (int)nX);
    cvt_w_kernel<<<dim3((unsigned)(nW / 256), 1, 3), dim3(256), 0, stream>>>(Wq, Wk, Wv, Wt);

    // Stage 2: QKV projections.  Grid: (row tiles, col tiles, {Q,K,V}).
    proj_kernel<<<dim3(BATCH * SEQ / 32, D_OUT / 64, 3), dim3(32), 0, stream>>>(
        Xh, Wt, Qh, Kh, Vt);

    // Stage 3: flash attention.  Grid: (q tiles, batch), 8 waves each.
    attn_kernel<<<dim3(SEQ / 16, BATCH), dim3(256), 0, stream>>>(Qh, Kh, Vt, out);
}
